// UniAxialMultiHeadAttentionBlock_81114752353012
// MI455X (gfx1250) — matrix operand order repaired and checked by reading, not compiled, not run on hardware
//
#include <hip/hip_runtime.h>
#include <hip/hip_bf16.h>

// ---------------------------------------------------------------------------
// UniAxialMultiHeadAttentionBlock for MI455X (gfx1250, wave32, WMMA + TDM)
// B=4 T=512 M=12 D=128 P=128 H=4 E=32 ENC=128
// ---------------------------------------------------------------------------

typedef _Float16 v8h  __attribute__((ext_vector_type(8)));
typedef _Float16 v16h __attribute__((ext_vector_type(16)));
typedef float    v8f  __attribute__((ext_vector_type(8)));

#define Bc   4
#define Tc   512
#define Mc   12
#define Dc   128
#define Pc   128
#define Hc   4
#define Ec   32
#define ENCc 128

#if defined(__HIP_DEVICE_COMPILE__) && __has_builtin(__builtin_amdgcn_tensor_load_to_lds) && __has_builtin(__builtin_amdgcn_s_wait_tensorcnt)
#define USE_TDM 1
#else
#define USE_TDM 0
#endif

__device__ __forceinline__ v8f wmma_f16(v16h a, v16h b, v8f c) {
  // D = A(16x32 f16) * B(32x16 f16) + C(16x16 f32)
  return __builtin_amdgcn_wmma_f32_16x16x32_f16(
      false, a, false, b, (short)0, c, false, false);
}

// A-fragment (16x32, f16) from row-major [row][k] buffer (LDS or global).
__device__ __forceinline__ v16h load_a(const _Float16* base, int stride, int kbase) {
  const int lane = threadIdx.x & 31;
  const _Float16* p = base + (lane & 15) * stride + kbase + ((lane >> 4) << 4);
  v8h lo = *(const v8h*)(p);
  v8h hi = *(const v8h*)(p + 8);
  v16h r;
#pragma unroll
  for (int i = 0; i < 8; ++i) { r[i] = lo[i]; r[i + 8] = hi[i]; }
  return r;
}

// B-fragment (32x16, f16) from B^T stored row-major [n][k]; one 32B load/lane.
__device__ __forceinline__ v16h load_bt(const _Float16* base, int stride,
                                        int nbase, int kbase) {
  const int lane = threadIdx.x & 31;
  const _Float16* p = base + (nbase + (lane & 15)) * stride + kbase + ((lane >> 4) << 4);
  return *(const v16h*)(p);
}

// Row reductions across the 16-lane half-groups (matches C-frag row layout).
__device__ __forceinline__ float grpmax(float v) {
#pragma unroll
  for (int m = 1; m <= 8; m <<= 1) v = fmaxf(v, __shfl_xor(v, m, 32));
  return v;
}
__device__ __forceinline__ float grpsum(float v) {
#pragma unroll
  for (int m = 1; m <= 8; m <<= 1) v += __shfl_xor(v, m, 32);
  return v;
}

#if USE_TDM
// ---------------------------------------------------------------------------
// Tensor Data Mover: async 2D tile load, 64 rows x 32 f16 elems (row stride
// 32 elems in memory), written to LDS with 64B data + 64B pad per row so the
// LDS row stride is 64 halves (interleaves the [k | k_t] halves of Ka).
// D# layout per CDNA5 ISA ch. 8.
// ---------------------------------------------------------------------------
typedef unsigned int tdm_u32x4 __attribute__((ext_vector_type(4)));
typedef int          tdm_i32x4 __attribute__((ext_vector_type(4)));
typedef int          tdm_i32x8 __attribute__((ext_vector_type(8)));

__device__ __forceinline__ void tdm_load_tile64x32(unsigned lds_byte_addr,
                                                   const _Float16* gsrc) {
  unsigned long long ga = (unsigned long long)(size_t)(const void*)gsrc;
  tdm_u32x4 g0;
  g0[0] = 1u;                                  // count=1 (valid), user mode
  g0[1] = lds_byte_addr;                       // LDS destination (bytes)
  g0[2] = (unsigned)ga;                        // global_addr[31:0]
  g0[3] = (unsigned)((ga >> 32) & 0x01ffffffu) | 0x80000000u; // addr[56:32]|type=2
  tdm_i32x8 g1;
  g1[0] = (1 << 16)      // data_size = 2 bytes
        | (1 << 20)      // pad_enable
        | (3 << 22)      // pad_interval: every 16 DWORDs (=64B row)
        | (15 << 25);    // pad_amount: 16 DWORDs (=64B)
  g1[1] = (int)(32u << 16);          // tensor_dim0[15:0] = 32 elems
  g1[2] = 0;                         // tensor_dim0 hi / tensor_dim1 lo
  g1[3] = (int)(16u | (32u << 16));  // tensor_dim1 = 1<<20 (hi16) | tile_dim0 = 32
  g1[4] = 64;                        // tile_dim1 = 64 rows, tile_dim2 = 0
  g1[5] = 32;                        // tensor_dim0_stride = 32 elems
  g1[6] = 0;
  g1[7] = 0;
  tdm_i32x4 z4 = {};
#if defined(__clang_major__) && __clang_major__ >= 23
  tdm_i32x8 z8 = {};
  __builtin_amdgcn_tensor_load_to_lds(g0, g1, z4, z4, z8, 0);
#else
  __builtin_amdgcn_tensor_load_to_lds(g0, g1, z4, z4, 0);
#endif
}
#endif  // USE_TDM

// ---------------------------------------------------------------------------
// Kernel 0: transpose + f16-convert all nine 128x128 weights. WT[w][p][d].
// ---------------------------------------------------------------------------
__global__ __launch_bounds__(256)
void k_prep_w(const float* w0, const float* w1, const float* w2,
              const float* w3, const float* w4, const float* w5,
              const float* w6, const float* w7, const float* w8,
              _Float16* __restrict__ wt) {
  int idx = blockIdx.x * 256 + threadIdx.x;        // 9*16384 total
  int w = idx >> 14, r = idx & 16383;
  int d = r >> 7, p = r & 127;
  const float* s;
  switch (w) {
    case 0: s = w0; break; case 1: s = w1; break; case 2: s = w2; break;
    case 3: s = w3; break; case 4: s = w4; break; case 5: s = w5; break;
    case 6: s = w6; break; case 7: s = w7; break; default: s = w8; break;
  }
  wt[(w << 14) + p * 128 + d] = (_Float16)s[d * 128 + p];
}

// ---------------------------------------------------------------------------
// Kernel 1: stage-1 projections. X(24576x128) @ {Wq,Wk,Wv} -> scattered
// (B,M,H,T,E) f16 tiles; 1/(2*sqrt(E)) folded into Q.
// ---------------------------------------------------------------------------
__global__ __launch_bounds__(128)
void k_proj1(const float* __restrict__ inp, const _Float16* __restrict__ wt,
             const float* __restrict__ bq, const float* __restrict__ bk,
             const float* __restrict__ bv,
             _Float16* __restrict__ Qh, _Float16* __restrict__ Kh,
             _Float16* __restrict__ Vh) {
  __shared__ alignas(32) _Float16 Xs[16][Dc];
  const int tile = blockIdx.x, sel = blockIdx.y;
  const int tid = threadIdx.x, wave = tid >> 5, lane = tid & 31;

  const float* xrow = inp + (size_t)tile * 16 * Dc;
  for (int idx = tid; idx < 16 * Dc; idx += 128)
    Xs[idx >> 7][idx & 127] = (_Float16)xrow[idx];
  __syncthreads();

  v16h a[4];
#pragma unroll
  for (int kc = 0; kc < 4; ++kc) a[kc] = load_a(&Xs[0][0], Dc, kc * 32);

  const _Float16* W = wt + sel * 16384;
  const float* bias = sel == 0 ? bq : (sel == 1 ? bk : bv);
  _Float16* out     = sel == 0 ? Qh : (sel == 1 ? Kh : Vh);
  const float scale = sel == 0 ? 0.08838834765f : 1.0f;  // 1/(2*sqrt(32))
  const int rofs = (lane >> 4) << 3;

  int gm[8], gt[8], gb[8];
#pragma unroll
  for (int i = 0; i < 8; ++i) {
    int g = tile * 16 + i + rofs;
    gm[i] = g % Mc; int q = g / Mc; gt[i] = q % Tc; gb[i] = q / Tc;
  }
#pragma unroll
  for (int c2 = 0; c2 < 2; ++c2) {
    int ct = wave * 2 + c2;
    int col = ct * 16 + (lane & 15);
    float bcol = bias[col];
    v8f c;
#pragma unroll
    for (int i = 0; i < 8; ++i) c[i] = bcol;
#pragma unroll
    for (int kc = 0; kc < 4; ++kc)
      c = wmma_f16(a[kc], load_bt(W, Dc, ct * 16, kc * 32), c);
    int h = col >> 5, e = col & 31;
#pragma unroll
    for (int i = 0; i < 8; ++i) {
      size_t addr = ((((size_t)gb[i] * Mc + gm[i]) * Hc + h) * Tc + gt[i]) * Ec + e;
      out[addr] = (_Float16)(c[i] * scale);
    }
  }
}

// ---------------------------------------------------------------------------
// Kernel 2: positional projections. pos(2048x128) @ {Wqt,Wkt} -> (B,H,T,E).
// ---------------------------------------------------------------------------
__global__ __launch_bounds__(128)
void k_projpos(const float* __restrict__ pos, const _Float16* __restrict__ wt,
               const float* __restrict__ bqt, const float* __restrict__ bkt,
               _Float16* __restrict__ Qth, _Float16* __restrict__ Kth) {
  __shared__ alignas(32) _Float16 Xs[16][Dc];
  const int tile = blockIdx.x, sel = blockIdx.y;
  const int tid = threadIdx.x, wave = tid >> 5, lane = tid & 31;

  const float* xrow = pos + (size_t)tile * 16 * Dc;
  for (int idx = tid; idx < 16 * Dc; idx += 128)
    Xs[idx >> 7][idx & 127] = (_Float16)xrow[idx];
  __syncthreads();

  v16h a[4];
#pragma unroll
  for (int kc = 0; kc < 4; ++kc) a[kc] = load_a(&Xs[0][0], Dc, kc * 32);

  const _Float16* W = wt + (3 + sel) * 16384;
  const float* bias = sel == 0 ? bqt : bkt;
  _Float16* out     = sel == 0 ? Qth : Kth;
  const float scale = sel == 0 ? 0.08838834765f : 1.0f;
  const int rofs = (lane >> 4) << 3;

#pragma unroll
  for (int c2 = 0; c2 < 2; ++c2) {
    int ct = wave * 2 + c2;
    int col = ct * 16 + (lane & 15);
    float bcol = bias[col];
    v8f c;
#pragma unroll
    for (int i = 0; i < 8; ++i) c[i] = bcol;
#pragma unroll
    for (int kc = 0; kc < 4; ++kc)
      c = wmma_f16(a[kc], load_bt(W, Dc, ct * 16, kc * 32), c);
    int h = col >> 5, e = col & 31;
#pragma unroll
    for (int i = 0; i < 8; ++i) {
      int g = tile * 16 + i + rofs;          // g = b*T + t
      int t = g & (Tc - 1), b = g >> 9;
      out[(((size_t)b * Hc + h) * Tc + t) * Ec + e] = (_Float16)(c[i] * scale);
    }
  }
}

// ---------------------------------------------------------------------------
// Kernel 3: stage-1 flash attention, augmented head dim 64 = [q|q_t],[k|k_t].
// Grid: (T/64, B*M*H). 4 waves, 16 query rows each. K/Kt tiles come in via
// double-buffered TDM async tile loads overlapped with compute.
// ---------------------------------------------------------------------------
__global__ __launch_bounds__(128)
void k_attn1(const _Float16* __restrict__ Qh, const _Float16* __restrict__ Kh,
             const _Float16* __restrict__ Vh, const _Float16* __restrict__ Qth,
             const _Float16* __restrict__ Kth,
             const unsigned char* __restrict__ mask, float* __restrict__ out1) {
  __shared__ alignas(32) _Float16 Qa[64][64];
  __shared__ alignas(32) _Float16 Ka[2][64][64];   // [k | k_t] per row
  __shared__ alignas(32) _Float16 VT[2][32][64];   // V transposed [e][s]
  __shared__ alignas(32) _Float16 Pl[4][16][64];
  __shared__ unsigned char Mk[2][64];

  const int bmh = blockIdx.y;
  const int b = bmh / (Mc * Hc);
  const int rm = bmh % (Mc * Hc);
  const int m = rm >> 2, h = rm & 3;
  const int q0 = blockIdx.x * 64;
  const int tid = threadIdx.x, wave = tid >> 5, lane = tid & 31;
  const int rofs = (lane >> 4) << 3;

  const size_t bh = (((size_t)b * Mc + m) * Hc + h) * Tc;
  const _Float16* Qb  = Qh  + bh * Ec;
  const _Float16* Kb  = Kh  + bh * Ec;
  const _Float16* Vb  = Vh  + bh * Ec;
  const _Float16* Qtb = Qth + (((size_t)b * Hc + h) * Tc) * Ec;
  const _Float16* Ktb = Kth + (((size_t)b * Hc + h) * Tc) * Ec;

  // ---- prologue: prefetch tile 0 into buffer 0, load Q tile --------------
#if USE_TDM
  if (wave == 0) {
    unsigned ka = (unsigned)(size_t)(void*)&Ka[0][0][0];
    tdm_load_tile64x32(ka,       Kb);
    tdm_load_tile64x32(ka + 64u, Ktb);
  }
#else
  for (int idx = tid; idx < 64 * 4; idx += 128) {
    int r = idx >> 2, c8 = (idx & 3) * 8;
    *(v8h*)&Ka[0][r][c8]      = *(const v8h*)&Kb [r * Ec + c8];
    *(v8h*)&Ka[0][r][32 + c8] = *(const v8h*)&Ktb[r * Ec + c8];
  }
#endif
  for (int idx = tid; idx < 64 * 32; idx += 128) {
    int s = idx >> 5, e = idx & 31;
    VT[0][e][s] = Vb[s * Ec + e];
  }
  if (tid < 64) Mk[0][tid] = mask[((size_t)b * Tc + tid) * Mc + m];
  for (int idx = tid; idx < 64 * 4; idx += 128) {
    int r = idx >> 2, c8 = (idx & 3) * 8;
    *(v8h*)&Qa[r][c8]      = *(const v8h*)&Qb [(q0 + r) * Ec + c8];
    *(v8h*)&Qa[r][32 + c8] = *(const v8h*)&Qtb[(q0 + r) * Ec + c8];
  }
#if USE_TDM
  if (wave == 0) __builtin_amdgcn_s_wait_tensorcnt(0);
#endif
  __syncthreads();

  v16h aq[2];
  aq[0] = load_a(&Qa[wave * 16][0], 64, 0);
  aq[1] = load_a(&Qa[wave * 16][0], 64, 32);

  float rmax[8], rsum[8];
  v8f o0 = {}, o1 = {};
#pragma unroll
  for (int i = 0; i < 8; ++i) { rmax[i] = -1e30f; rsum[i] = 0.f; }

  for (int j = 0; j < 8; ++j) {
    const int buf = j & 1, nb = buf ^ 1;

    // ---- prefetch tile j+1 into the other buffer (async) -----------------
    if (j < 7) {
      const int s1 = (j + 1) * 64;
#if USE_TDM
      if (wave == 0) {
        unsigned ka = (unsigned)(size_t)(void*)&Ka[nb][0][0];
        tdm_load_tile64x32(ka,       Kb  + (size_t)s1 * Ec);
        tdm_load_tile64x32(ka + 64u, Ktb + (size_t)s1 * Ec);
      }
#else
      for (int idx = tid; idx < 64 * 4; idx += 128) {
        int r = idx >> 2, c8 = (idx & 3) * 8;
        *(v8h*)&Ka[nb][r][c8]      = *(const v8h*)&Kb [(s1 + r) * Ec + c8];
        *(v8h*)&Ka[nb][r][32 + c8] = *(const v8h*)&Ktb[(s1 + r) * Ec + c8];
      }
#endif
      for (int idx = tid; idx < 64 * 32; idx += 128) {
        int s = idx >> 5, e = idx & 31;
        VT[nb][e][s] = Vb[(s1 + s) * Ec + e];
      }
      if (tid < 64) Mk[nb][tid] = mask[((size_t)b * Tc + s1 + tid) * Mc + m];
    }

    // ---- S = Qaug @ Kaug^T on buffer `buf` --------------------------------
    float pv[4][8], bm[8];
#pragma unroll
    for (int i = 0; i < 8; ++i) bm[i] = -1e30f;
#pragma unroll
    for (int nt = 0; nt < 4; ++nt) {
      v8f s = {};
      s = wmma_f16(aq[0], load_bt(&Ka[buf][0][0], 64, nt * 16, 0),  s);
      s = wmma_f16(aq[1], load_bt(&Ka[buf][0][0], 64, nt * 16, 32), s);
      bool mk = Mk[buf][nt * 16 + (lane & 15)] != 0;
#pragma unroll
      for (int i = 0; i < 8; ++i) {
        float v = mk ? s[i] : -1e30f;
        pv[nt][i] = v;
        bm[i] = fmaxf(bm[i], v);
      }
    }
    float nmax[8], alpha[8];
#pragma unroll
    for (int i = 0; i < 8; ++i) {
      float g = grpmax(bm[i]);
      nmax[i] = fmaxf(rmax[i], g);
      alpha[i] = __expf(rmax[i] - nmax[i]);
      rmax[i] = nmax[i];
    }
    float ls[8];
#pragma unroll
    for (int i = 0; i < 8; ++i) ls[i] = 0.f;
#pragma unroll
    for (int nt = 0; nt < 4; ++nt) {
      bool mk = Mk[buf][nt * 16 + (lane & 15)] != 0;
#pragma unroll
      for (int i = 0; i < 8; ++i) {
        float p = mk ? __expf(pv[nt][i] - nmax[i]) : 0.f;
        ls[i] += p;
        Pl[wave][i + rofs][nt * 16 + (lane & 15)] = (_Float16)p;
      }
    }
#pragma unroll
    for (int i = 0; i < 8; ++i) {
      rsum[i] = rsum[i] * alpha[i] + grpsum(ls[i]);
      o0[i] *= alpha[i];
      o1[i] *= alpha[i];
    }
    __syncthreads();
    v16h ap0 = load_a(&Pl[wave][0][0], 64, 0);
    v16h ap1 = load_a(&Pl[wave][0][0], 64, 32);
    o0 = wmma_f16(ap0, load_bt(&VT[buf][0][0], 64, 0, 0),   o0);
    o0 = wmma_f16(ap1, load_bt(&VT[buf][0][0], 64, 0, 32),  o0);
    o1 = wmma_f16(ap0, load_bt(&VT[buf][0][0], 64, 16, 0),  o1);
    o1 = wmma_f16(ap1, load_bt(&VT[buf][0][0], 64, 16, 32), o1);

    // ---- publish tile j+1 (TDM done + LDS stores drained) -----------------
#if USE_TDM
    if (wave == 0) __builtin_amdgcn_s_wait_tensorcnt(0);
#endif
    __syncthreads();
  }

#pragma unroll
  for (int i = 0; i < 8; ++i) {
    float inv = rsum[i] > 0.f ? 1.0f / rsum[i] : 0.f;
    int t = q0 + wave * 16 + i + rofs;
    size_t base = (((size_t)b * Tc + t) * Mc + m) * Pc + h * Ec;
    out1[base + (lane & 15)]      = o0[i] * inv;
    out1[base + 16 + (lane & 15)] = o1[i] * inv;
  }
}

// ---------------------------------------------------------------------------
// Kernel 4: stage-2 fully fused per (b,t): q2/k2/v2 proj, 12x12 modality
// attention (wave == head), o2 @ Wo + bo. Grid: B*T blocks, 128 threads.
// ---------------------------------------------------------------------------
__global__ __launch_bounds__(128)
void k_attn2(const float* __restrict__ out1, const _Float16* __restrict__ wt,
             const float* __restrict__ bq2, const float* __restrict__ bk2,
             const float* __restrict__ bv2, const float* __restrict__ bo,
             const unsigned char* __restrict__ mask, float* __restrict__ out) {
  __shared__ alignas(32) _Float16 X [16][Pc];
  __shared__ alignas(32) _Float16 Q2[16][Pc];
  __shared__ alignas(32) _Float16 K2[16][Pc];
  __shared__ alignas(32) _Float16 V2T[Pc][32];   // [p][n_mod], cols 16..31 zero pad
  __shared__ alignas(32) _Float16 P2[4][16][32]; // per-head P, cols 12..31 zero
  __shared__ alignas(32) _Float16 O2[16][Pc];
  __shared__ unsigned char Mk[16];

  const int bt = blockIdx.x;
  const int tid = threadIdx.x, wave = tid >> 5, lane = tid & 31;
  const int rofs = (lane >> 4) << 3;

  const float* xg = out1 + (size_t)bt * Mc * Pc;
  for (int idx = tid; idx < 16 * Pc; idx += 128) {
    int r = idx >> 7, c = idx & 127;
    X[r][c] = (_Float16)(r < Mc ? xg[r * Pc + c] : 0.f);
  }
  for (int idx = tid; idx < Pc * 32; idx += 128) (&V2T[0][0])[idx] = (_Float16)0.f;
  for (int idx = tid; idx < 4 * 16 * 32; idx += 128) (&P2[0][0][0])[idx] = (_Float16)0.f;
  if (tid < 16) Mk[tid] = (tid < Mc) ? mask[(size_t)bt * Mc + tid] : (unsigned char)0;
  __syncthreads();

  v16h ax[4];
#pragma unroll
  for (int kc = 0; kc < 4; ++kc) ax[kc] = load_a(&X[0][0], Pc, kc * 32);

  const float inv_se = 0.17677669529f;  // 1/sqrt(32), folded into q2
#pragma unroll
  for (int c2 = 0; c2 < 2; ++c2) {
    int ct = wave * 2 + c2;
    int col = ct * 16 + (lane & 15);
    {
      float bcol = bq2[col];
      v8f c;
#pragma unroll
      for (int i = 0; i < 8; ++i) c[i] = bcol;
      const _Float16* W = wt + 5 * 16384;
#pragma unroll
      for (int kc = 0; kc < 4; ++kc)
        c = wmma_f16(ax[kc], load_bt(W, Pc, ct * 16, kc * 32), c);
#pragma unroll
      for (int i = 0; i < 8; ++i) Q2[i + rofs][col] = (_Float16)(c[i] * inv_se);
    }
    {
      float bcol = bk2[col];
      v8f c;
#pragma unroll
      for (int i = 0; i < 8; ++i) c[i] = bcol;
      const _Float16* W = wt + 6 * 16384;
#pragma unroll
      for (int kc = 0; kc < 4; ++kc)
        c = wmma_f16(ax[kc], load_bt(W, Pc, ct * 16, kc * 32), c);
#pragma unroll
      for (int i = 0; i < 8; ++i) K2[i + rofs][col] = (_Float16)c[i];
    }
    {
      float bcol = bv2[col];
      v8f c;
#pragma unroll
      for (int i = 0; i < 8; ++i) c[i] = bcol;
      const _Float16* W = wt + 7 * 16384;
#pragma unroll
      for (int kc = 0; kc < 4; ++kc)
        c = wmma_f16(ax[kc], load_bt(W, Pc, ct * 16, kc * 32), c);
#pragma unroll
      for (int i = 0; i < 8; ++i) V2T[col][i + rofs] = (_Float16)c[i];  // transpose
    }
  }
  __syncthreads();

  // per-head 16x16 attention; wave == head
  const int h = wave;
  v8f zf = {};
  v8f s = wmma_f16(load_a(&Q2[0][0], Pc, h * 32),
                   load_bt(&K2[0][0], Pc, 0, h * 32), zf);
  bool mk = Mk[lane & 15] != 0;
  float sv[8], rmax[8], rsum[8];
#pragma unroll
  for (int i = 0; i < 8; ++i) sv[i] = mk ? s[i] : -1e30f;
#pragma unroll
  for (int i = 0; i < 8; ++i) rmax[i] = grpmax(sv[i]);
#pragma unroll
  for (int i = 0; i < 8; ++i) sv[i] = mk ? __expf(sv[i] - rmax[i]) : 0.f;
#pragma unroll
  for (int i = 0; i < 8; ++i) rsum[i] = grpsum(sv[i]);
#pragma unroll
  for (int i = 0; i < 8; ++i) P2[h][i + rofs][lane & 15] = (_Float16)sv[i];
  __syncthreads();

  v16h apv = load_a(&P2[h][0][0], 32, 0);
#pragma unroll
  for (int et = 0; et < 2; ++et) {
    v8f o = wmma_f16(apv, load_bt(&V2T[0][0], 32, h * 32 + et * 16, 0), zf);
#pragma unroll
    for (int i = 0; i < 8; ++i) {
      float inv = rsum[i] > 0.f ? 1.f / rsum[i] : 0.f;
      O2[i + rofs][h * 32 + et * 16 + (lane & 15)] = (_Float16)(o[i] * inv);
    }
  }
  __syncthreads();

  // o2 @ Wo + bo
  v16h ao[4];
#pragma unroll
  for (int kc = 0; kc < 4; ++kc) ao[kc] = load_a(&O2[0][0], Pc, kc * 32);
#pragma unroll
  for (int c2 = 0; c2 < 2; ++c2) {
    int ct = wave * 2 + c2;
    int col = ct * 16 + (lane & 15);
    float bcol = bo[col];
    v8f c;
#pragma unroll
    for (int i = 0; i < 8; ++i) c[i] = bcol;
    const _Float16* W = wt + 8 * 16384;
#pragma unroll
    for (int kc = 0; kc < 4; ++kc)
      c = wmma_f16(ao[kc], load_bt(W, Pc, ct * 16, kc * 32), c);
#pragma unroll
    for (int i = 0; i < 8; ++i) {
      int mrow = i + rofs;
      if (mrow < Mc)
        out[((size_t)bt * Mc + mrow) * ENCc + col] = c[i];
    }
  }
}

// ---------------------------------------------------------------------------
extern "C" void kernel_launch(void* const* d_in, const int* in_sizes, int n_in,
                              void* d_out, int out_size, void* d_ws, size_t ws_size,
                              hipStream_t stream) {
  (void)in_sizes; (void)n_in; (void)out_size; (void)ws_size;
  const float* inp  = (const float*)d_in[0];
  const float* pos  = (const float*)d_in[1];
  const unsigned char* mask = (const unsigned char*)d_in[2];
  const float* Wq  = (const float*)d_in[3];  const float* bq  = (const float*)d_in[4];
  const float* Wk  = (const float*)d_in[5];  const float* bk  = (const float*)d_in[6];
  const float* Wv  = (const float*)d_in[7];  const float* bv  = (const float*)d_in[8];
  const float* Wqt = (const float*)d_in[9];  const float* bqt = (const float*)d_in[10];
  const float* Wkt = (const float*)d_in[11]; const float* bkt = (const float*)d_in[12];
  const float* Wq2 = (const float*)d_in[13]; const float* bq2 = (const float*)d_in[14];
  const float* Wk2 = (const float*)d_in[15]; const float* bk2 = (const float*)d_in[16];
  const float* Wv2 = (const float*)d_in[17]; const float* bv2 = (const float*)d_in[18];
  const float* Wo  = (const float*)d_in[19]; const float* bo  = (const float*)d_in[20];
  float* outp = (float*)d_out;

  char* ws = (char*)d_ws;
  const size_t szWT = 9u * 16384u * sizeof(_Float16);                    // 294912
  const size_t szQ  = (size_t)Bc * Mc * Hc * Tc * Ec * sizeof(_Float16); // 6291456
  const size_t szQt = (size_t)Bc * Hc * Tc * Ec * sizeof(_Float16);      // 524288
  _Float16* WT  = (_Float16*)(ws);
  _Float16* Qh  = (_Float16*)(ws + szWT);
  _Float16* Kh  = (_Float16*)(ws + szWT + szQ);
  _Float16* Vh  = (_Float16*)(ws + szWT + 2 * szQ);
  _Float16* Qth = (_Float16*)(ws + szWT + 3 * szQ);
  _Float16* Kth = (_Float16*)(ws + szWT + 3 * szQ + szQt);
  float*    O1  = (float*)   (ws + szWT + 3 * szQ + 2 * szQt);

  // 0) weight transpose + f16 convert (order: Wq Wk Wv Wqt Wkt Wq2 Wk2 Wv2 Wo)
  k_prep_w<<<(9 * 16384) / 256, 256, 0, stream>>>(Wq, Wk, Wv, Wqt, Wkt,
                                                  Wq2, Wk2, Wv2, Wo, WT);
  // 1) stage-1 projections: 24576 rows / 16 = 1536 tiles x {q,k,v}
  k_proj1<<<dim3(1536, 3), 128, 0, stream>>>(inp, WT, bq, bk, bv, Qh, Kh, Vh);
  // 2) positional projections: 2048 rows / 16 = 128 tiles x {qt,kt}
  k_projpos<<<dim3(128, 2), 128, 0, stream>>>(pos, WT, bqt, bkt, Qth, Kth);
  // 3) stage-1 flash attention: (T/64) x (B*M*H), TDM double-buffered K tiles
  k_attn1<<<dim3(Tc / 64, Bc * Mc * Hc), 128, 0, stream>>>(Qh, Kh, Vh, Qth, Kth,
                                                           mask, O1);
  // 4) stage-2 fused modality attention + output projection: B*T blocks
  k_attn2<<<dim3(Bc * Tc), 128, 0, stream>>>(O1, WT, bq2, bk2, bv2, bo, mask, outp);
}
